// VQ_StraightThrough_72473278152748
// MI455X (gfx1250) — matrix operand order repaired and checked by reading, not compiled, not run on hardware
//
#include <hip/hip_runtime.h>

typedef __attribute__((ext_vector_type(16))) _Float16 v16h;
typedef __attribute__((ext_vector_type(8)))  float    v8f;
typedef __attribute__((ext_vector_type(4)))  unsigned int v4u;
typedef __attribute__((ext_vector_type(8)))  int      v8i;
typedef __attribute__((ext_vector_type(4)))  int      v4i;

#define NB 32
#define DD 64
#define HH 64
#define WW 64
#define NK 512
#define NTOK      (NB*HH*WW)        // 131072 tokens
#define TOK_PER_B (HH*WW)           // 4096
#define ZSTRIDE_B (DD*TOK_PER_B)    // 262144 floats per batch image

// ---- workspace layout (bytes) ----
#define WS_IDX_OFF   0                          // 131072 int32 indices
#define WS_BFRAG_OFF (NTOK*4)                   // 2048 lanes * 16 halves (f16 B fragments) = 64KB
#define WS_CNORM_OFF (WS_BFRAG_OFF + 2048*16*2) // 512 f32 ||c||^2 (contiguous after frags)
#define STAGE_DWORDS (16384 + 512)              // 66KB staged: frags + cnorm

// ---- output layout (floats), concatenated in reference return order ----
#define OUT_Z_OFF    0
#define OUT_LOSS_OFF (NB*DD*HH*WW)      // 8388608
#define OUT_IND_OFF  (OUT_LOSS_OFF + NB)// 8388640

// ---------------------------------------------------------------------------
// Kernel 0: pack codebook into f16 WMMA B-fragments, compute ||c||^2, zero loss
// B fragment layout (16-bit B 32x16): lane l holds column N = l&15,
// K = j + 16*(l>>4) for half j (k-step adds +32).
// ---------------------------------------------------------------------------
__global__ void __launch_bounds__(256)
vq_prep(const float* __restrict__ cb, float* __restrict__ outF, void* ws)
{
    int tid = blockIdx.x * 256 + threadIdx.x;
    _Float16* bfrag = (_Float16*)((char*)ws + WS_BFRAG_OFF);
    float*    cnorm = (float*)((char*)ws + WS_CNORM_OFF);

    if (tid < 2048) {                       // 32 tiles * 2 ksteps * 32 lanes
        int lane = tid & 31;
        int ts   = tid >> 5;                // t*2 + s
        int s    = ts & 1;
        int t    = ts >> 1;
        int n    = t * 16 + (lane & 15);
        int kbase = ((lane >> 4) ? 16 : 0) + 32 * s;
        const float* row = cb + n * DD;
        #pragma unroll
        for (int j = 0; j < 16; ++j)
            bfrag[tid * 16 + j] = (_Float16)row[kbase + j];
    } else if (tid < 2048 + NB) {
        outF[OUT_LOSS_OFF + (tid - 2048)] = 0.0f;   // zero loss accumulators
    } else if (tid < 2048 + NB + NK) {
        int n = tid - 2048 - NB;
        const float* row = cb + n * DD;
        float s = 0.0f;
        #pragma unroll
        for (int k = 0; k < DD; ++k) s += row[k] * row[k];
        cnorm[n] = s;
    }
}

// ---------------------------------------------------------------------------
// Kernel 1: per-wave 16-token tile against all 512 codes.
// B fragments + ||c||^2 (66KB) staged global->LDS once per block via the
// Tensor Data Mover (tensor_load_to_lds + s_wait_tensorcnt), then the hot
// loop is ds_load_b128 + v_wmma_f32_16x16x32_f16.
// ---------------------------------------------------------------------------
__global__ void __launch_bounds__(256)
vq_assign(const float* __restrict__ z, float* __restrict__ outF, void* ws)
{
    __shared__ unsigned int lds_stage[STAGE_DWORDS];  // 64KB frags + 2KB cnorm
    __shared__ float lds_val[8][16][16];
    __shared__ int   lds_idx[8][16][16];
    __shared__ float lds_loss;

    int wave = threadIdx.x >> 5;
    int lane = threadIdx.x & 31;
    int g    = lane >> 4;       // half-wave group
    int l15  = lane & 15;

    if (threadIdx.x == 0) lds_loss = 0.0f;

    // ---- stage codebook fragments + cnorm into LDS via the TDM ----
#if defined(__gfx1250__) && __has_builtin(__builtin_amdgcn_tensor_load_to_lds)
    if (wave == 0) {
        unsigned long long ga = (unsigned long long)((const char*)ws + WS_BFRAG_OFF);
        unsigned int ldsOff = (unsigned int)(unsigned long long)(void*)&lds_stage[0];
        // D# group0: count=1 | lds_addr | global_addr[56:0] | type=2
        v4u g0 = { 1u, ldsOff,
                   (unsigned int)(ga & 0xFFFFFFFFu),
                   (unsigned int)(((ga >> 32) & 0x01FFFFFFu) | 0x80000000u) };
        // D# group1: data_size=4B; tensor_dim0 = tile_dim0 = STAGE_DWORDS (1-D tile)
        v8i g1 = { 0x00020000,
                   (int)((STAGE_DWORDS & 0xFFFF) << 16),     // tensor_dim0 lo16
                   0x00010000,                               // tensor_dim1 = 1
                   (int)((STAGE_DWORDS & 0xFFFF) << 16),     // tile_dim0
                   0,                                        // tile_dim1/2 = 0
                   STAGE_DWORDS,                             // tensor_dim0_stride lo32
                   0, 0 };
        v4i gz4 = { 0, 0, 0, 0 };
        v8i gz8 = { 0, 0, 0, 0, 0, 0, 0, 0 };
        // 6-arg toolchain form: (g0, g1, g2, g3, g4, cpol)
        __builtin_amdgcn_tensor_load_to_lds(g0, g1, gz4, gz4, gz8, 0);
        __builtin_amdgcn_s_wait_tensorcnt(0);
    }
#else
    {   // cooperative staging fallback
        const unsigned int* src = (const unsigned int*)((const char*)ws + WS_BFRAG_OFF);
        for (int i = threadIdx.x; i < STAGE_DWORDS; i += 256)
            lds_stage[i] = src[i];
    }
#endif
    __syncthreads();

    const v16h*  bfragL = (const v16h*)&lds_stage[0];
    const float* cnormL = (const float*)&lds_stage[16384];
    int*         idxOut = (int*)ws;

    int T       = blockIdx.x * 8 + wave;   // 16-token tile id
    int tokBase = T * 16;
    int tok     = tokBase + l15;           // this lane's token row (A-matrix M)
    int b       = tok >> 12;               // 4096 tokens per batch image
    int zbase   = b * ZSTRIDE_B + (tok & (TOK_PER_B - 1));

    // --- load A fragments (f32 -> f16) and accumulate ||z||^2 in f32 ---
    // A 16x32 f16 layout: K = j + 16*g, k-step 1 adds +32.
    float sumsq = 0.0f;
    v16h a0, a1;
    #pragma unroll
    for (int j = 0; j < 16; ++j) {
        int K = j + 16 * g;
        float v0 = z[zbase + K * TOK_PER_B];
        float v1 = z[zbase + (K + 32) * TOK_PER_B];
        sumsq += v0 * v0 + v1 * v1;
        a0[j] = (_Float16)v0;
        a1[j] = (_Float16)v1;
    }
    // lane l and l+16 split one token's features: pair-sum gives full ||z||^2
    float sumsq_tok = sumsq + __shfl_xor(sumsq, 16, 32);

    float bestV[8];
    int   bestI[8];
    #pragma unroll
    for (int r = 0; r < 8; ++r) { bestV[r] = -3.4e38f; bestI[r] = 0; }

    #pragma unroll 2
    for (int t = 0; t < 32; ++t) {         // 32 tiles of 16 codes
        float bias = -0.5f * cnormL[t * 16 + l15];
        v8f acc = {bias, bias, bias, bias, bias, bias, bias, bias};
        v16h b0 = bfragL[(t * 2 + 0) * 32 + lane];
        v16h b1 = bfragL[(t * 2 + 1) * 32 + lane];
        acc = __builtin_amdgcn_wmma_f32_16x16x32_f16(false, a0, false, b0,
                                                     (short)0, acc, false, false);
        acc = __builtin_amdgcn_wmma_f32_16x16x32_f16(false, a1, false, b1,
                                                     (short)0, acc, false, false);
        int code = t * 16 + l15;
        #pragma unroll
        for (int r = 0; r < 8; ++r) {
            float v = acc[r];
            if (v > bestV[r]) { bestV[r] = v; bestI[r] = code; }  // ties keep smaller idx
        }
    }

    // scatter per-lane candidates: row = token m, col = candidate lane
    #pragma unroll
    for (int r = 0; r < 8; ++r) {
        int m = r + 8 * g;
        lds_val[wave][m][l15] = bestV[r];
        lds_idx[wave][m][l15] = bestI[r];
    }
    __syncthreads();

    if (lane < 16) {
        float bv = lds_val[wave][lane][0];
        int   bi = lds_idx[wave][lane][0];
        #pragma unroll
        for (int i = 1; i < 16; ++i) {
            float v = lds_val[wave][lane][i];
            int   x = lds_idx[wave][lane][i];
            if (v > bv || (v == bv && x < bi)) { bv = v; bi = x; }
        }
        int tok2 = tokBase + lane;
        idxOut[tok2] = bi;
        outF[OUT_IND_OFF + tok2] = (float)bi;
        float d2 = sumsq_tok - 2.0f * bv;     // ||z - c_best||^2
        d2 += __shfl_xor(d2, 1, 32);
        d2 += __shfl_xor(d2, 2, 32);
        d2 += __shfl_xor(d2, 4, 32);
        d2 += __shfl_xor(d2, 8, 32);
        if (lane == 0) atomicAdd(&lds_loss, d2);
    }
    __syncthreads();
    if (threadIdx.x == 0) atomicAdd(&outF[OUT_LOSS_OFF + b], lds_loss);
}

// ---------------------------------------------------------------------------
// Kernel 2: gather z_q = codebook[ind] and write NCHW output with coalesced
// stores (stage selected rows in LDS, padded to kill bank conflicts).
// Block 0 also applies the loss scale 1.25/(H*W*D).
// ---------------------------------------------------------------------------
__global__ void __launch_bounds__(256)
vq_gather(const float* __restrict__ cb, float* __restrict__ outF, const void* ws)
{
    __shared__ int   sInd[64];
    __shared__ float srow[64][65];
    const int* idxBuf = (const int*)ws;

    int tid = threadIdx.x;
    int b   = blockIdx.x >> 6;
    int h   = blockIdx.x & 63;

    if (tid < 64) sInd[tid] = idxBuf[b * TOK_PER_B + h * WW + tid];
    __syncthreads();

    // coalesced row fetches: each 64-thread chunk pulls one 256B codebook row
    for (int i = tid; i < 64 * 64; i += 256) {
        int ts = i >> 6;
        int d  = i & 63;
        srow[ts][d] = cb[sInd[ts] * DD + d];
    }
    __syncthreads();

    int w = tid & 63;
    int q = tid >> 6;
    float* obase = outF + b * ZSTRIDE_B + h * WW + w;   // out[b, d, h, w]
    #pragma unroll
    for (int d = q; d < 64; d += 4)
        obase[d * TOK_PER_B] = srow[w][d];              // 128B coalesced stores

    if (blockIdx.x == 0 && tid < NB)
        outF[OUT_LOSS_OFF + tid] *= 1.25f / 262144.0f;
}

// ---------------------------------------------------------------------------
extern "C" void kernel_launch(void* const* d_in, const int* in_sizes, int n_in,
                              void* d_out, int out_size, void* d_ws, size_t ws_size,
                              hipStream_t stream)
{
    const float* z  = (const float*)d_in[0];   // [32,64,64,64] f32 NCHW
    const float* cb = (const float*)d_in[1];   // [512,64] f32
    float* outF = (float*)d_out;

    vq_prep  <<<11,       256, 0, stream>>>(cb, outF, d_ws);
    vq_assign<<<NTOK/128, 256, 0, stream>>>(z,  outF, d_ws);
    vq_gather<<<NB*HH,    256, 0, stream>>>(cb, outF, d_ws);
}
